// EQConv_73254962200774
// MI455X (gfx1250) — compile-verified, matrix operand order checked by reading, not run on hardware
//
#include <hip/hip_runtime.h>
#include <hip/hip_bf16.h>

typedef __bf16 v16bf __attribute__((ext_vector_type(16)));
typedef float  v8f   __attribute__((ext_vector_type(8)));
typedef int    v4i   __attribute__((ext_vector_type(4)));

typedef __attribute__((address_space(1))) v4i as1_v4i;   // global (__device__)
typedef __attribute__((address_space(3))) v4i as3_v4i;   // LDS (__shared__)

#define B_FRAGS     96                 // 6 N-tiles x 16 K-steps
#define FRAG_STRIDE 1088               // 1024B fragment + 64B bank-conflict skew
#define B_CHUNKS    (B_FRAGS * 64)     // 16-byte chunks in packed global B
#define BS_BYTES    (B_FRAGS * FRAG_STRIDE)   // 104448

#if defined(__HIP_DEVICE_COMPILE__) && \
    __has_builtin(__builtin_amdgcn_global_load_async_to_lds_b128) && \
    __has_builtin(__builtin_amdgcn_s_wait_asynccnt)
#define USE_ASYNC_LDS 1
#else
#define USE_ASYNC_LDS 0
#endif

// ---------------------------------------------------------------------------
// Kernel 0: zero the output accumulator (harness poisons d_out with 0xAA).
// ---------------------------------------------------------------------------
__global__ void zero_out_kernel(float* __restrict__ out, int n) {
  for (int i = blockIdx.x * blockDim.x + threadIdx.x; i < n;
       i += gridDim.x * blockDim.x)
    out[i] = 0.0f;
}

// ---------------------------------------------------------------------------
// Kernel 1: pack W2 (fp32, 16 x 3072) * (1/sqrt(16)) into bf16 WMMA B-fragments
// in global scratch. B is (K=512) x (N=96); fragment (j,s) is a 32x16 bf16
// block laid out so lane l's 16 K-values are 32 contiguous bytes matching the
// 16-bit operand layout (lanes 0-15: K 0-7/16-23, lanes 16-31: K 8-15/24-31,
// 2 values per dword). Row k_local of step s is (c=s, u=k_local); column
// n = kg*32+o maps to W2[c, kg*1024 + u*32 + o].
// ---------------------------------------------------------------------------
__global__ void pack_b_kernel(const float* __restrict__ W2,
                              __bf16* __restrict__ B) {
  int tid = blockIdx.x * blockDim.x + threadIdx.x;
  if (tid >= B_FRAGS * 32 * 8) return;          // 24576 dword-pairs
  int r    = tid & 7;                            // vgpr slot 0..7
  int l    = (tid >> 3) & 31;                    // lane
  int s    = (tid >> 8) & 15;                    // K-step
  int j    = tid >> 12;                          // N-tile
  int half = l >> 4;
  int n    = j * 16 + (l & 15);
  int kg   = n >> 5;
  int o    = n & 31;
  int u0   = ((r >> 2) << 4) + half * 8 + ((r & 3) << 1);
  float x0 = W2[s * 3072 + kg * 1024 + (u0    ) * 32 + o] * 0.25f;
  float x1 = W2[s * 3072 + kg * 1024 + (u0 + 1) * 32 + o] * 0.25f;
  int idx  = (j * 16 + s) * 512 + l * 16 + r * 2;   // __bf16 units (packed)
  B[idx]     = (__bf16)x0;
  B[idx + 1] = (__bf16)x1;
}

// ---------------------------------------------------------------------------
// Kernel 2: fused eqconv. One wave = one tile of 16 edges.
//   B is staged once per block into LDS (async global->LDS), with a
//   16B-per-8-lanes skew so inner-loop ds_load_b128 is conflict-free.
//   GEMM: (16 x 512) @ (512 x 96) via 96 v_wmma_f32_16x16x32_bf16; the
//   A-fragment of K-step s is the outer product h[:,s] (x) f[:,:] built in
//   registers. K-loop fully unrolled so hreg[] indexing is static (no
//   v_movrels/M0). Epilogue applies sh1/sh2 and scatter-adds 288 floats per
//   edge with global_atomic_add_f32.
// ---------------------------------------------------------------------------
__global__ void __launch_bounds__(256)
eqconv_kernel(const float* __restrict__ f_in, const float* __restrict__ pos,
              const float* __restrict__ W1,  const int* __restrict__ esrc,
              const int* __restrict__ edst,  const int* __restrict__ mrp,
              const __bf16* __restrict__ Bg, float* __restrict__ out,
              int E, int nTiles) {
  __shared__ __attribute__((aligned(16))) char BsArr[BS_BYTES];
  __shared__ float W1S[160];          // 10 x 16
  __shared__ float hS[8][16][16];     // per-wave: h per edge
  __shared__ float shS[8][16][8];     // per-wave: sh1(3)+sh2(5) per edge
  __shared__ int   srcS[8][16];
  __shared__ int   dstS[8][16];

  const int t    = threadIdx.x;
  const int lane = t & 31;
  const int w    = t >> 5;
  const int tile = blockIdx.x * 8 + w;
  const bool active = tile < nTiles;
  const int m16  = lane & 15;
  const int half = lane >> 4;

  // ---- stage packed B into LDS with bank-conflict skew ----
  // global chunk i (16B): frag = i>>6, lane l = (i&63)>>1, hh = i&1
  // LDS offset = frag*1088 + l*32 + (l>>3)*16 + hh*16
  for (int i = t; i < B_CHUNKS; i += 256) {
    int frag = i >> 6;
    int rem  = i & 63;
    int l    = rem >> 1;
    int hh   = rem & 1;
    int ldsOff = frag * FRAG_STRIDE + l * 32 + ((l >> 3) << 4) + (hh << 4);
#if USE_ASYNC_LDS
    __builtin_amdgcn_global_load_async_to_lds_b128(
        (as1_v4i*)((const char*)Bg + (size_t)i * 16),
        (as3_v4i*)(BsArr + ldsOff), 0, 0);
#else
    *(int4*)(BsArr + ldsOff) = ((const int4*)Bg)[i];
#endif
  }
#if USE_ASYNC_LDS
  __builtin_amdgcn_s_wait_asynccnt(0);
#endif

  if (t < 160) W1S[t] = W1[t];

  // ---- phase A: geometry + radial basis (no W1 dependence) ----
  float embv[10];
  float validf = 0.0f;
  if (active && lane < 16) {
    int  e     = tile * 16 + m16;
    bool valid = e < E;
    validf     = valid ? 1.0f : 0.0f;
    int src = valid ? esrc[e] : 0;
    int dst = valid ? edst[e] : 0;
    srcS[w][m16] = src;
    dstS[w][m16] = valid ? dst : -1;

    float ax = pos[src * 3 + 0], ay = pos[src * 3 + 1], az = pos[src * 3 + 2];
    float bx = pos[dst * 3 + 0], by = pos[dst * 3 + 1], bz = pos[dst * 3 + 2];
    float vx = bx - ax, vy = by - ay, vz = bz - az;
    float rr  = sqrtf(vx * vx + vy * vy + vz * vz);
    float inv = 1.0f / fmaxf(rr, 1e-12f);
    float ux = vx * inv, uy = vy * inv, uz = vz * inv;

    const float c3  = 1.7320508075688772f;
    const float c5  = 2.23606797749979f;
    const float c15 = 3.872983346207417f;
    shS[w][m16][0] = c3 * ux;
    shS[w][m16][1] = c3 * uy;
    shS[w][m16][2] = c3 * uz;
    shS[w][m16][3] = c15 * ux * uz;
    shS[w][m16][4] = c15 * ux * uy;
    shS[w][m16][5] = c5 * (uy * uy - 0.5f * (ux * ux + uz * uz));
    shS[w][m16][6] = c15 * uy * uz;
    shS[w][m16][7] = 0.5f * c15 * (uz * uz - ux * ux);

    // soft-one-hot radial basis; sqrt(NUM_BASIS) cancels against W1/sqrt(NB)
    float mr   = (float)mrp[0];
    float step = mr * (1.0f / 11.0f);
    float rs   = rr / step;
#pragma unroll
    for (int b = 0; b < 10; ++b) {
      float d  = rs - (float)(b + 1);
      float e1 = d + 1.0f, e2 = 1.0f - d;
      float v  = 0.0f;
      if (e1 > 0.0f && e2 > 0.0f)
        v = 8.4335741f * __expf(-1.0f / e1 - 1.0f / e2);  // 1.14136*e^2*sus*sus
      embv[b] = v;
    }
  }

  __syncthreads();  // W1S and BsArr visible block-wide

  // ---- phase B: h = sqrt(2)*relu(emb @ W1) ----
  if (active && lane < 16) {
    float hh[16];
#pragma unroll
    for (int c = 0; c < 16; ++c) hh[c] = 0.0f;
#pragma unroll
    for (int b = 0; b < 10; ++b) {
      float eb = embv[b];
#pragma unroll
      for (int c = 0; c < 16; ++c) hh[c] += eb * W1S[b * 16 + c];
    }
#pragma unroll
    for (int c = 0; c < 16; ++c)
      hS[w][m16][c] = validf * 1.41421356f * fmaxf(hh[c], 0.0f);
  }
  // hS/shS/srcS written and read by the same wave: DS ops are in-order.

  if (!active) return;

  // ---- per-lane f slice: u = {half*8 .. +7, 16+half*8 .. +7} ----
  int src = srcS[w][m16];
  const float* fp = f_in + (size_t)src * 32 + half * 8;
  float4 f0 = *(const float4*)(fp);
  float4 f1 = *(const float4*)(fp + 4);
  float4 f2 = *(const float4*)(fp + 16);
  float4 f3 = *(const float4*)(fp + 20);
  float fv[16] = {f0.x, f0.y, f0.z, f0.w, f1.x, f1.y, f1.z, f1.w,
                  f2.x, f2.y, f2.z, f2.w, f3.x, f3.y, f3.z, f3.w};

  float hreg[16];
#pragma unroll
  for (int i = 0; i < 16; ++i) hreg[i] = hS[w][m16][i];

  v8f acc[6] = {};
  // lane's swizzled 32B slot base within each fragment
  const char* Bl = BsArr + lane * 32 + ((lane >> 3) << 4);

#pragma unroll
  for (int s = 0; s < 16; ++s) {
    float hs = hreg[s];
    v16bf a;
#pragma unroll
    for (int i = 0; i < 16; ++i) a[i] = (__bf16)(hs * fv[i]);
#pragma unroll
    for (int j = 0; j < 6; ++j) {
      const int4* p = (const int4*)(Bl + (j * 16 + s) * FRAG_STRIDE);
      union { int4 q[2]; v16bf v; } u;
      u.q[0] = p[0];
      u.q[1] = p[1];
      acc[j] = __builtin_amdgcn_wmma_f32_16x16x32_bf16(
          false, a, false, u.v, (short)0, acc[j], false, false);
    }
  }

  // ---- epilogue: scale, multiply by sh, scatter-add ----
  // scale = 1/sqrt(MUL) * 1/sqrt(avg_num_neighbors=10) = 1/sqrt(320)
  const float scale = 0.055901699437494745f;
#pragma unroll
  for (int v = 0; v < 8; ++v) {
    int mm = half * 8 + v;        // edge row (C layout: lanes>=16 hold M=v+8)
    int dd = dstS[w][mm];
    if (dd >= 0) {
      float* op = out + (size_t)dd * 288;
      float x0 = acc[0][v] * scale;   // n = m16     -> out0
      float x1 = acc[1][v] * scale;   // n = 16+m16  -> out0
      __hip_atomic_fetch_add(op + m16,      x0, __ATOMIC_RELAXED, __HIP_MEMORY_SCOPE_AGENT);
      __hip_atomic_fetch_add(op + 16 + m16, x1, __ATOMIC_RELAXED, __HIP_MEMORY_SCOPE_AGENT);
      float x2 = acc[2][v] * scale;   // o = m16
      float x3 = acc[3][v] * scale;   // o = 16+m16
#pragma unroll
      for (int q = 0; q < 3; ++q) {
        float sv = shS[w][mm][q];
        __hip_atomic_fetch_add(op + 32 + m16 * 3 + q,        x2 * sv, __ATOMIC_RELAXED, __HIP_MEMORY_SCOPE_AGENT);
        __hip_atomic_fetch_add(op + 32 + (16 + m16) * 3 + q, x3 * sv, __ATOMIC_RELAXED, __HIP_MEMORY_SCOPE_AGENT);
      }
      float x4 = acc[4][v] * scale;
      float x5 = acc[5][v] * scale;
#pragma unroll
      for (int q = 0; q < 5; ++q) {
        float sv = shS[w][mm][3 + q];
        __hip_atomic_fetch_add(op + 128 + m16 * 5 + q,        x4 * sv, __ATOMIC_RELAXED, __HIP_MEMORY_SCOPE_AGENT);
        __hip_atomic_fetch_add(op + 128 + (16 + m16) * 5 + q, x5 * sv, __ATOMIC_RELAXED, __HIP_MEMORY_SCOPE_AGENT);
      }
    }
  }
}

// ---------------------------------------------------------------------------
extern "C" void kernel_launch(void* const* d_in, const int* in_sizes, int n_in,
                              void* d_out, int out_size, void* d_ws, size_t ws_size,
                              hipStream_t stream) {
  const float* f_in = (const float*)d_in[0];
  const float* pos  = (const float*)d_in[1];
  const float* W1   = (const float*)d_in[2];
  const float* W2   = (const float*)d_in[3];
  const int*   esrc = (const int*)d_in[4];
  const int*   edst = (const int*)d_in[5];
  const int*   mrp  = (const int*)d_in[6];
  float*  out = (float*)d_out;
  __bf16* B   = (__bf16*)d_ws;      // 512 x 96 bf16 = 98304 bytes

  const int E = in_sizes[4];

  int zb = (out_size + 255) / 256;
  if (zb > 2048) zb = 2048;
  zero_out_kernel<<<zb, 256, 0, stream>>>(out, out_size);

  pack_b_kernel<<<(B_FRAGS * 32 * 8 + 255) / 256, 256, 0, stream>>>(W2, B);

  const int nTiles = (E + 15) / 16;
  const int blocks = (nTiles + 7) / 8;   // 8 waves (tiles) per 256-thread block
  eqconv_kernel<<<blocks, 256, 0, stream>>>(f_in, pos, W1, esrc, edst, mrp,
                                            B, out, E, nTiles);
}
